// TopKMoE_24885040513423
// MI455X (gfx1250) — compile-verified, matrix operand order checked by reading, not run on hardware
//
#include <hip/hip_runtime.h>
#include <hip/hip_bf16.h>
#include <math.h>

// Shapes from the reference
#define D_MODEL 1024
#define D_FF    4096
#define N_EXP   8
#define T_TOK   4096   // 2 * 2048 tokens
#define TM      32     // tokens per workgroup tile (two WMMA M tiles)

typedef __attribute__((ext_vector_type(16))) __bf16 v16bf;
typedef __attribute__((ext_vector_type(8)))  float  v8f;

union FragBF {
    v16bf v;
    uint4 q[2];
};

__device__ __forceinline__ unsigned short f2bf(float f) {
    unsigned int u = __float_as_uint(f);
    unsigned int r = u + 0x7FFFu + ((u >> 16) & 1u);   // round-to-nearest-even
    return (unsigned short)(r >> 16);
}

__device__ __forceinline__ float gelu_exact(float v) {
    return 0.5f * v * (1.0f + erff(v * 0.70710678118654752440f));
}

// ---------------------------------------------------------------------------
// Zero output accumulator and per-expert counters
// ---------------------------------------------------------------------------
__global__ void init_kernel(float* __restrict__ out, int* __restrict__ cnt) {
    int i = blockIdx.x * 256 + threadIdx.x;
    if (i < T_TOK * D_MODEL) out[i] = 0.0f;
    if (i < N_EXP) cnt[i] = 0;
}

// ---------------------------------------------------------------------------
// x fp32 -> bf16 (same layout); activations then stream from L2 as bf16
// ---------------------------------------------------------------------------
__global__ void convert_x_kernel(const float* __restrict__ x,
                                 unsigned short* __restrict__ xb) {
    int i = blockIdx.x * 256 + threadIdx.x;   // T_TOK*D_MODEL = 4M, exact grid
    xb[i] = f2bf(x[i]);
}

// ---------------------------------------------------------------------------
// fp32 -> bf16 + transpose so WMMA B-fragments are contiguous in K.
// w1 : [E, D, F] row-major  ->  w1t : [E, F, D]  (n-major, k contiguous)
// ---------------------------------------------------------------------------
__global__ void convert_w1_kernel(const float* __restrict__ w1,
                                  unsigned short* __restrict__ w1t) {
    size_t idx = (size_t)blockIdx.x * 256 + threadIdx.x;        // over E*F*D
    size_t per = (size_t)D_FF * D_MODEL;
    size_t e = idx / per;
    size_t r = idx - e * per;
    size_t n = r / D_MODEL;          // 0..4095
    size_t k = r - n * D_MODEL;      // 0..1023
    w1t[idx] = f2bf(w1[e * (size_t)D_MODEL * D_FF + k * D_FF + n]);
}

// w2 : [E, F, D] row-major  ->  w2t : [E, D, F]
__global__ void convert_w2_kernel(const float* __restrict__ w2,
                                  unsigned short* __restrict__ w2t) {
    size_t idx = (size_t)blockIdx.x * 256 + threadIdx.x;        // over E*D*F
    size_t per = (size_t)D_MODEL * D_FF;
    size_t e = idx / per;
    size_t r = idx - e * per;
    size_t n = r / D_FF;             // 0..1023
    size_t k = r - n * D_FF;         // 0..4095
    w2t[idx] = f2bf(w2[e * (size_t)D_FF * D_MODEL + k * D_MODEL + n]);
}

// ---------------------------------------------------------------------------
// Router: logits = x @ rw + rb, top-2, softmax gates, bucket tokens per expert
// ---------------------------------------------------------------------------
__global__ void router_kernel(const float* __restrict__ x,
                              const float* __restrict__ rw,
                              const float* __restrict__ rb,
                              int* __restrict__ cnt,
                              int* __restrict__ etok,
                              float* __restrict__ egate) {
    int t = blockIdx.x * 256 + threadIdx.x;
    if (t >= T_TOK) return;
    float acc[N_EXP];
#pragma unroll
    for (int e = 0; e < N_EXP; ++e) acc[e] = rb[e];
    const float* xr = x + (size_t)t * D_MODEL;
    for (int k = 0; k < D_MODEL; ++k) {
        float xv = xr[k];
        const float* r = rw + k * N_EXP;
#pragma unroll
        for (int e = 0; e < N_EXP; ++e) acc[e] = fmaf(xv, r[e], acc[e]);
    }
    // top-2, lowest index wins ties (matches jax.lax.top_k)
    int i0 = 0; float m0 = acc[0];
#pragma unroll
    for (int e = 1; e < N_EXP; ++e) if (acc[e] > m0) { m0 = acc[e]; i0 = e; }
    int i1 = -1; float m1 = -3.4e38f;
#pragma unroll
    for (int e = 0; e < N_EXP; ++e)
        if (e != i0 && acc[e] > m1) { m1 = acc[e]; i1 = e; }
    float e1 = expf(m1 - m0);
    float inv = 1.0f / (1.0f + e1);
    float g0 = inv, g1 = e1 * inv;

    int s0 = atomicAdd(&cnt[i0], 1);
    etok [i0 * T_TOK + s0] = t;
    egate[i0 * T_TOK + s0] = g0;
    int s1 = atomicAdd(&cnt[i1], 1);
    etok [i1 * T_TOK + s1] = t;
    egate[i1 * T_TOK + s1] = g1;
}

// ---------------------------------------------------------------------------
// Fused expert MLP over a 32-token tile of one expert.
//   Phase 1: H = gelu(Xg @ W1 + b1)  -> held entirely in LDS (32x4096 bf16)
//   Phase 2: Y = H @ W2 + b2, scatter gate*Y with global_atomic_add_f32
// 8 waves / 256 threads; bf16 WMMA 16x16x32, f32 accumulate.
// Per k-step: 2 A-frags + 4 B-frags -> 8 WMMAs (B reused across both M tiles).
// LDS: Hs 256KB + tile metadata (CDNA5 WGP has 320KB).
// ---------------------------------------------------------------------------
__global__ __launch_bounds__(256) void moe_mlp_kernel(
    const unsigned short* __restrict__ xb,
    const float* __restrict__ b1,
    const float* __restrict__ b2,
    const unsigned short* __restrict__ w1t,
    const unsigned short* __restrict__ w2t,
    const int* __restrict__ cnt,
    const int* __restrict__ etok,
    const float* __restrict__ egate,
    float* __restrict__ out) {
    extern __shared__ char smem[];
    unsigned short* Hs = (unsigned short*)smem;                 // 32*4096 bf16
    int*   Tid = (int*)  (smem + TM * D_FF * 2);                // 32 ids
    float* Gs  = (float*)(smem + TM * D_FF * 2 + 128);          // 32 gates

    const int e      = blockIdx.x >> 7;     // 128 tiles per expert
    const int tile   = blockIdx.x & 127;
    const int count  = cnt[e];
    const int tstart = tile * TM;
    if (tstart >= count) return;            // uniform early-exit

    const int tid  = threadIdx.x;
    const int lane = tid & 31;
    const int wv   = tid >> 5;

    if (tid < TM) {
        int g = tstart + tid;
        if (g < count) {
            Tid[tid] = etok [e * T_TOK + g];
            Gs [tid] = egate[e * T_TOK + g];
        } else {                            // padded row: gate 0 kills it
            Tid[tid] = 0;
            Gs [tid] = 0.0f;
        }
    }
    __syncthreads();

    // Per-lane fragment geometry (ISA 7.12.2, 16-bit A/B layout)
    const int koff  = ((lane >> 4) & 1) * 8;  // K sub-offset per lane half
    const int ncol  = lane & 15;              // B/C/D column within tile
    const int mrow  = lane & 15;              // A row within M tile
    const int mbase = (lane >> 4) * 8;        // C/D row base

    // Gathered bf16 activation row base for this lane's A rows (both M tiles)
    const unsigned short* arow0 = xb + (size_t)Tid[mrow]      * D_MODEL;
    const unsigned short* arow1 = xb + (size_t)Tid[16 + mrow] * D_MODEL;

    // ---------------- Phase 1: H = gelu(X @ W1 + b1) ----------------
    const unsigned short* w1e = w1t + (size_t)e * D_FF * D_MODEL;
    const float* b1e = b1 + e * D_FF;
    for (int nb = 0; nb < 8; ++nb) {          // wave covers 512 cols of D_FF
        const int nbase = wv * 512 + nb * 64;
        v8f c[2][4];
#pragma unroll
        for (int jt = 0; jt < 4; ++jt) {
            float bv = b1e[nbase + jt * 16 + ncol];
            c[0][jt] = (v8f){bv, bv, bv, bv, bv, bv, bv, bv};
            c[1][jt] = c[0][jt];
        }
        for (int k0 = 0; k0 < D_MODEL; k0 += 32) {
            FragBF a0, a1;
            a0.q[0] = *(const uint4*)(arow0 + k0 + koff);
            a0.q[1] = *(const uint4*)(arow0 + k0 + koff + 16);
            a1.q[0] = *(const uint4*)(arow1 + k0 + koff);
            a1.q[1] = *(const uint4*)(arow1 + k0 + koff + 16);
#pragma unroll
            for (int jt = 0; jt < 4; ++jt) {
                const int n = nbase + jt * 16 + ncol;
                FragBF b;
                const unsigned short* pb = w1e + (size_t)n * D_MODEL + k0 + koff;
                b.q[0] = *(const uint4*)pb;
                b.q[1] = *(const uint4*)(pb + 16);
                c[0][jt] = __builtin_amdgcn_wmma_f32_16x16x32_bf16(
                    false, a0.v, false, b.v, (short)0, c[0][jt], false, false);
                c[1][jt] = __builtin_amdgcn_wmma_f32_16x16x32_bf16(
                    false, a1.v, false, b.v, (short)0, c[1][jt], false, false);
            }
        }
#pragma unroll
        for (int mt = 0; mt < 2; ++mt) {
#pragma unroll
            for (int jt = 0; jt < 4; ++jt) {
                const int n = nbase + jt * 16 + ncol;
#pragma unroll
                for (int i = 0; i < 8; ++i) {
                    Hs[(mt * 16 + mbase + i) * D_FF + n] = f2bf(gelu_exact(c[mt][jt][i]));
                }
            }
        }
    }
    __syncthreads();

    // ---------------- Phase 2: Y = H @ W2 + b2, scatter ----------------
    const unsigned short* w2e = w2t + (size_t)e * D_MODEL * D_FF;
    const float* b2e = b2 + e * D_MODEL;
    for (int nb = 0; nb < 2; ++nb) {          // wave covers 128 cols of D_MODEL
        const int nbase = wv * 128 + nb * 64;
        v8f c[2][4];
#pragma unroll
        for (int jt = 0; jt < 4; ++jt) {
            float bv = b2e[nbase + jt * 16 + ncol];
            c[0][jt] = (v8f){bv, bv, bv, bv, bv, bv, bv, bv};
            c[1][jt] = c[0][jt];
        }
        for (int k0 = 0; k0 < D_FF; k0 += 32) {
            FragBF a0, a1;
            const unsigned short* pa0 = Hs + mrow * D_FF + k0 + koff;
            const unsigned short* pa1 = Hs + (16 + mrow) * D_FF + k0 + koff;
            a0.q[0] = *(const uint4*)pa0;
            a0.q[1] = *(const uint4*)(pa0 + 16);
            a1.q[0] = *(const uint4*)pa1;
            a1.q[1] = *(const uint4*)(pa1 + 16);
#pragma unroll
            for (int jt = 0; jt < 4; ++jt) {
                const int n = nbase + jt * 16 + ncol;
                FragBF b;
                const unsigned short* pb = w2e + (size_t)n * D_FF + k0 + koff;
                b.q[0] = *(const uint4*)pb;
                b.q[1] = *(const uint4*)(pb + 16);
                c[0][jt] = __builtin_amdgcn_wmma_f32_16x16x32_bf16(
                    false, a0.v, false, b.v, (short)0, c[0][jt], false, false);
                c[1][jt] = __builtin_amdgcn_wmma_f32_16x16x32_bf16(
                    false, a1.v, false, b.v, (short)0, c[1][jt], false, false);
            }
        }
#pragma unroll
        for (int mt = 0; mt < 2; ++mt) {
#pragma unroll
            for (int jt = 0; jt < 4; ++jt) {
                const int n = nbase + jt * 16 + ncol;
#pragma unroll
                for (int i = 0; i < 8; ++i) {
                    const int m = mt * 16 + mbase + i;
                    atomicAdd(out + (size_t)Tid[m] * D_MODEL + n, Gs[m] * c[mt][jt][i]);
                }
            }
        }
    }
}

// ---------------------------------------------------------------------------
extern "C" void kernel_launch(void* const* d_in, const int* in_sizes, int n_in,
                              void* d_out, int out_size, void* d_ws, size_t ws_size,
                              hipStream_t stream) {
    const float* x  = (const float*)d_in[0];
    const float* rw = (const float*)d_in[1];
    const float* rb = (const float*)d_in[2];
    const float* w1 = (const float*)d_in[3];
    const float* b1 = (const float*)d_in[4];
    const float* w2 = (const float*)d_in[5];
    const float* b2 = (const float*)d_in[6];
    float* out = (float*)d_out;

    // Workspace layout
    char* ws = (char*)d_ws;
    int*   cnt   = (int*)ws;                               // 8 ints (padded 256B)
    int*   etok  = (int*)(ws + 256);                       // E*T ids   (128KB)
    float* egate = (float*)(ws + 256 + 131072);            // E*T gates (128KB)
    unsigned short* w1t = (unsigned short*)(ws + 256 + 262144);
    unsigned short* w2t = w1t + (size_t)N_EXP * D_FF * D_MODEL;   // +64MB
    unsigned short* xb  = w2t + (size_t)N_EXP * D_FF * D_MODEL;   // +64MB (8MB)

    // 1) zero accumulator output + counters
    init_kernel<<<(T_TOK * D_MODEL + 255) / 256, 256, 0, stream>>>(out, cnt);

    // 2) one-time fp32->bf16 of activations + transposed expert weights
    convert_x_kernel<<<(T_TOK * D_MODEL) / 256, 256, 0, stream>>>(x, xb);
    const int nconv = (int)(((size_t)N_EXP * D_FF * D_MODEL) / 256);   // 131072
    convert_w1_kernel<<<nconv, 256, 0, stream>>>(w1, w1t);
    convert_w2_kernel<<<nconv, 256, 0, stream>>>(w2, w2t);

    // 3) route tokens -> per-expert buckets with softmaxed top-2 gates
    router_kernel<<<T_TOK / 256, 256, 0, stream>>>(x, rw, rb, cnt, etok, egate);

    // 4) fused expert MLP, 32-token tiles, bf16 WMMA, H kept in LDS
    size_t smem = (size_t)TM * D_FF * 2 + 256;
    moe_mlp_kernel<<<dim3(N_EXP * (T_TOK / TM)), 256, smem, stream>>>(
        xb, b1, b2, w1t, w2t, cnt, etok, egate, out);
}
